// LoraLinear_24781961298482
// MI455X (gfx1250) — matrix operand order repaired and checked by reading, not compiled, not run on hardware
//
#include <hip/hip_runtime.h>

// ---------------------------------------------------------------------------
// LoRA linear:  out = x @ W^T + 2.0 * (x @ A^T) @ B^T
//   x: [64, 4096] f32   W: [16384, 4096] f32
//   A: [64, 4096] f32   B: [16384, 64] f32   out: [64, 16384] f32
//
// Roofline: weight stream = 256 MB -> ~11 us at 23.3 TB/s; needs ~780 TFLOPS
// effective, so fp32 is converted in-register to bf16 and fed to
// v_wmma_f32_16x16x32_bf16. W is streamed with NT cache hints so the 192 MB
// L2 stays dedicated to x (re-read by every block), B, and out.
// ---------------------------------------------------------------------------

#define TOK   64
#define INF   4096
#define OUTF  16384
#define RANK  64
#define KT    64            // K-tile per staging iteration
#define NIT   (INF / KT)    // 64 iterations
#define LDSU  36            // uints per LDS row (= 72 bf16, padded from 64)

typedef __attribute__((ext_vector_type(16))) __bf16 v16bf;
typedef __attribute__((ext_vector_type(8)))  float  v8f;
typedef __attribute__((ext_vector_type(4)))  float  v4f;   // native clang vector (NT-load capable)
typedef __attribute__((ext_vector_type(2)))  float  v2f;
typedef __attribute__((ext_vector_type(2)))  __bf16 v2bf;

union Frag {
    unsigned int u[8];
    uint4        u4[2];
    v16bf        v;
};

// packed f32x2 -> bf16x2 (RNE); expected to lower to v_cvt_pk_bf16_f32
__device__ __forceinline__ unsigned int pack2_bf16(float lo, float hi) {
    v2f f; f[0] = lo; f[1] = hi;
    v2bf h = __builtin_convertvector(f, v2bf);
    union { v2bf h; unsigned int u; } cv;
    cv.h = h;
    return cv.u;
}

// ---------------------------------------------------------------------------
// Kernel 1: xa[t][r] = 2.0 * dot(x[t,:], A[r,:])   -> d_ws  (64x64 f32)
// ---------------------------------------------------------------------------
__global__ __launch_bounds__(64)
void lora_xa_kernel(const float* __restrict__ x, const float* __restrict__ A,
                    float* __restrict__ xa) {
    const int t = blockIdx.x;      // token
    const int r = threadIdx.x;     // rank index
    const v4f* xr = reinterpret_cast<const v4f*>(x + (size_t)t * INF);
    const v4f* ar = reinterpret_cast<const v4f*>(A + (size_t)r * INF);
    float s = 0.0f;
    #pragma unroll 4
    for (int k = 0; k < INF / 4; ++k) {
        v4f xv = xr[k];
        v4f av = ar[k];
        s += xv.x * av.x + xv.y * av.y + xv.z * av.z + xv.w * av.w;
    }
    xa[t * RANK + r] = 2.0f * s;
}

// ---------------------------------------------------------------------------
// Kernel 2: 64x64 output tile per block; 8 waves; wave -> 16x32 strip.
// Double-buffered LDS: global loads for tile kt+1 are in flight while the
// WMMAs for tile kt run; one barrier per iteration.
// ---------------------------------------------------------------------------
__global__ __launch_bounds__(256)
void lora_main_kernel(const float* __restrict__ x, const float* __restrict__ W,
                      const float* __restrict__ B, const float* __restrict__ xa,
                      float* __restrict__ out) {
    __shared__ unsigned int xs[2][TOK * LDSU];  // x tile, bf16-packed
    __shared__ unsigned int ws[2][64  * LDSU];  // W tile, bf16-packed

    const int tid  = threadIdx.x;
    const int lane = tid & 31;
    const int wave = tid >> 5;
    const int wm   = wave >> 1;     // 0..3 : 16-row strip
    const int wn   = wave & 1;      // 0..1 : 32-col half
    const int l15  = lane & 15;
    const int kh   = lane >> 4;     // half-wave K-group select
    const int colBase = blockIdx.x * 64;

    // staging decomposition: 256 threads = 16 rows x 16 float4s per pass
    const int srow = tid >> 4;      // 0..15
    const int scol = tid & 15;      // float4 column

    v8f acc0 = {};
    v8f acc1 = {};

    v4f xr[4], wr[4];

    // ---- prologue: fetch tile 0 and stage into buffer 0 ----
    #pragma unroll
    for (int p = 0; p < 4; ++p) {
        const int r = srow + p * 16;
        xr[p] = *reinterpret_cast<const v4f*>(x + (size_t)r * INF + scol * 4);
        wr[p] = __builtin_nontemporal_load(
                    reinterpret_cast<const v4f*>(W + (size_t)(colBase + r) * INF + scol * 4));
    }
    #pragma unroll
    for (int p = 0; p < 4; ++p) {
        const int r = srow + p * 16;
        uint2 xp; xp.x = pack2_bf16(xr[p].x, xr[p].y); xp.y = pack2_bf16(xr[p].z, xr[p].w);
        uint2 wp; wp.x = pack2_bf16(wr[p].x, wr[p].y); wp.y = pack2_bf16(wr[p].z, wr[p].w);
        *reinterpret_cast<uint2*>(&xs[0][r * LDSU + scol * 2]) = xp;
        *reinterpret_cast<uint2*>(&ws[0][r * LDSU + scol * 2]) = wp;
    }
    __syncthreads();

    for (int kt = 0; kt < NIT; ++kt) {
        const int cur = kt & 1;
        const bool more = (kt + 1) < NIT;

        // ---- issue global loads for the NEXT tile (overlap with WMMA) ----
        if (more) {
            const int kn = (kt + 1) * KT;
            #pragma unroll
            for (int p = 0; p < 4; ++p) {
                const int r = srow + p * 16;
                xr[p] = *reinterpret_cast<const v4f*>(x + (size_t)r * INF + kn + scol * 4);
                wr[p] = __builtin_nontemporal_load(
                            reinterpret_cast<const v4f*>(W + (size_t)(colBase + r) * INF + kn + scol * 4));
            }
        }

        // ---- two 16x16x32 WMMA K-steps over the CURRENT tile ----
        const unsigned int* xsb = xs[cur];
        const unsigned int* wsb = ws[cur];
        #pragma unroll
        for (int s = 0; s < 2; ++s) {
            Frag a;
            const int am = wm * 16 + l15;
            a.u4[0] = *reinterpret_cast<const uint4*>(&xsb[am * LDSU + s * 16 + kh * 8]);
            a.u4[1] = *reinterpret_cast<const uint4*>(&xsb[am * LDSU + s * 16 + kh * 8 + 4]);

            Frag b0, b1;
            const int o0 = wn * 32 + l15;
            const int o1 = o0 + 16;
            b0.u4[0] = *reinterpret_cast<const uint4*>(&wsb[o0 * LDSU + s * 16 + kh * 8]);
            b0.u4[1] = *reinterpret_cast<const uint4*>(&wsb[o0 * LDSU + s * 16 + kh * 8 + 4]);
            b1.u4[0] = *reinterpret_cast<const uint4*>(&wsb[o1 * LDSU + s * 16 + kh * 8]);
            b1.u4[1] = *reinterpret_cast<const uint4*>(&wsb[o1 * LDSU + s * 16 + kh * 8 + 4]);

            acc0 = __builtin_amdgcn_wmma_f32_16x16x32_bf16(false, a.v, false, b0.v,
                                                           (short)0, acc0, false, false);
            acc1 = __builtin_amdgcn_wmma_f32_16x16x32_bf16(false, a.v, false, b1.v,
                                                           (short)0, acc1, false, false);
        }

        // ---- convert + store NEXT tile into the other buffer ----
        if (more) {
            unsigned int* xsn = xs[cur ^ 1];
            unsigned int* wsn = ws[cur ^ 1];
            #pragma unroll
            for (int p = 0; p < 4; ++p) {
                const int r = srow + p * 16;
                uint2 xp; xp.x = pack2_bf16(xr[p].x, xr[p].y); xp.y = pack2_bf16(xr[p].z, xr[p].w);
                uint2 wp; wp.x = pack2_bf16(wr[p].x, wr[p].y); wp.y = pack2_bf16(wr[p].z, wr[p].w);
                *reinterpret_cast<uint2*>(&xsn[r * LDSU + scol * 2]) = xp;
                *reinterpret_cast<uint2*>(&wsn[r * LDSU + scol * 2]) = wp;
            }
        }
        __syncthreads();
    }

    // ---- LoRA correction: acc += (2*xa)[64x64] @ B^T, two K-steps of 32 ----
    {
        const int am  = wm * 16 + l15;
        const int ob0 = colBase + wn * 32 + l15;
        const int ob1 = ob0 + 16;
        #pragma unroll
        for (int s = 0; s < 2; ++s) {
            Frag a;
            #pragma unroll
            for (int j = 0; j < 4; ++j) {
                const int r0 = s * 32 + kh * 16 + 2 * j;
                v2f v0 = *reinterpret_cast<const v2f*>(xa + am * RANK + r0);
                v2f v1 = *reinterpret_cast<const v2f*>(xa + am * RANK + r0 + 8);
                a.u[j]     = pack2_bf16(v0.x, v0.y);
                a.u[j + 4] = pack2_bf16(v1.x, v1.y);
            }
            Frag b0, b1;
            #pragma unroll
            for (int j = 0; j < 8; ++j) {
                const int r0 = s * 32 + kh * 16 + 2 * j;
                v2f v0 = *reinterpret_cast<const v2f*>(B + (size_t)ob0 * RANK + r0);
                v2f v1 = *reinterpret_cast<const v2f*>(B + (size_t)ob1 * RANK + r0);
                b0.u[j] = pack2_bf16(v0.x, v0.y);
                b1.u[j] = pack2_bf16(v1.x, v1.y);
            }
            acc0 = __builtin_amdgcn_wmma_f32_16x16x32_bf16(false, a.v, false, b0.v,
                                                           (short)0, acc0, false, false);
            acc1 = __builtin_amdgcn_wmma_f32_16x16x32_bf16(false, a.v, false, b1.v,
                                                           (short)0, acc1, false, false);
        }
    }

    // ---- store C (streaming, never re-read): VGPR v -> M = wm*16+kh*8+v ----
    #pragma unroll
    for (int v = 0; v < 8; ++v) {
        const int row = wm * 16 + kh * 8 + v;
        __builtin_nontemporal_store(acc0[v], out + (size_t)row * OUTF + colBase + wn * 32 + l15);
        __builtin_nontemporal_store(acc1[v], out + (size_t)row * OUTF + colBase + wn * 32 + 16 + l15);
    }
}

// ---------------------------------------------------------------------------
extern "C" void kernel_launch(void* const* d_in, const int* in_sizes, int n_in,
                              void* d_out, int out_size, void* d_ws, size_t ws_size,
                              hipStream_t stream) {
    const float* x = (const float*)d_in[0];
    const float* W = (const float*)d_in[1];
    const float* A = (const float*)d_in[2];
    const float* B = (const float*)d_in[3];
    float* out = (float*)d_out;
    float* xa  = (float*)d_ws;   // 64*64 f32 = 16 KB scratch

    lora_xa_kernel<<<TOK, RANK, 0, stream>>>(x, A, xa);
    lora_main_kernel<<<OUTF / 64, 256, 0, stream>>>(x, W, B, xa, out);
}
